// FOGNN_58274116272547
// MI455X (gfx1250) — matrix operand order repaired and checked by reading, not compiled, not run on hardware
//
#include <hip/hip_runtime.h>

// ---------------------------------------------------------------------------
// Bipartite GNN conv (learn2branch-style) for gfx1250.
// Algebraic folding (exact):
//   * BN drops all first-linear biases.
//   * m_i / m_j are per-node with count-weighted BN stats.
//   * concat GEMMs split; BN affine folded into second linears.
// Final per-edge work: 3 chained bf16 WMMA GEMMs on edge features + gathers
// + relu + f32 atomics into agg. Memory-bound (~2.3GB => ~100us @ 23.3TB/s).
// This revision: weights stored pre-swizzled in WMMA B-fragment order
// (one 32B vector load per fragment) and LDS tiles padded (stride 68) to
// kill the 16-way bank conflict on A-fragment reads.
// ---------------------------------------------------------------------------

typedef __bf16 bf16;
typedef __attribute__((ext_vector_type(16))) __bf16 v16bf;
typedef __attribute__((ext_vector_type(8)))  float  v8f;

#define WAVES 8           // waves per block (256 threads, wave32)
#define TROWS 16          // rows per wave tile
#define TSTR  68          // padded LDS row stride (floats) -> conflict-free A reads

// Swizzled weight storage: fragment-major. For element (k, col) of a 64x64
// matrix, the owning fragment is (khalf = k>>5, nt = col>>4), lane encodes
// (col&15) plus the K-group bit, and e = k&15 is contiguous in memory.
__device__ __forceinline__ int swz_idx(int k, int col) {
  int khalf = k >> 5, hi = (k >> 4) & 1, e = k & 15;
  int nt = col >> 4, lane = (col & 15) | (hi << 4);
  return (((khalf * 4 + nt) * 32) + lane) * 16 + e;
}

// ---- WMMA helpers: 16x64 tile (rows x cols), K=64, A from LDS f32 tile -----

__device__ __forceinline__ v16bf a_from_lds(const float* t, int lane, int khalf) {
  v16bf a;
  int r    = lane & 15;
  int base = khalf * 32 + ((lane >> 4) & 1) * 16;
#pragma unroll
  for (int e = 0; e < 16; ++e) {
    int k = base + e;   // contiguous k = 16h + e (matches swizzled B)
    a[e] = (bf16)t[r * TSTR + k];
  }
  return a;
}

__device__ __forceinline__ v16bf b_from_w(const bf16* W, int lane, int khalf, int nt) {
  // pre-swizzled: one aligned 32-byte vector load
  return *(const v16bf*)(W + (((khalf * 4 + nt) * 32) + lane) * 16);
}

// acc[4] covers the full 16x64 output tile (4 N-tiles of 16)
__device__ __forceinline__ void gemm_tile(const float* t, const bf16* W, int lane, v8f acc[4]) {
  v16bf a0 = a_from_lds(t, lane, 0);
  v16bf a1 = a_from_lds(t, lane, 1);
#pragma unroll
  for (int nt = 0; nt < 4; ++nt) {
    acc[nt] = __builtin_amdgcn_wmma_f32_16x16x32_bf16(
        false, a0, false, b_from_w(W, lane, 0, nt), (short)0, acc[nt], false, false);
    acc[nt] = __builtin_amdgcn_wmma_f32_16x16x32_bf16(
        false, a1, false, b_from_w(W, lane, 1, nt), (short)0, acc[nt], false, false);
  }
}

// D layout: VGPR v -> row v (+8 for lanes>=16), col = (lane&15)+nt*16
__device__ __forceinline__ void d_to_lds(float* t, const v8f acc[4], int lane,
                                         const float* bias, int relu) {
  int hi = (lane >> 4) & 1, c0 = lane & 15;
#pragma unroll
  for (int nt = 0; nt < 4; ++nt) {
    int col = c0 + nt * 16;
    float b = bias ? bias[col] : 0.f;
#pragma unroll
    for (int v = 0; v < 8; ++v) {
      float x = acc[nt][v] + b;
      if (relu) x = fmaxf(x, 0.f);
      t[(v + hi * 8) * TSTR + col] = x;
    }
  }
}

__device__ __forceinline__ void stage_raw(float* t, const float* X, int row0, int n, int lane) {
  for (int i = lane; i < TROWS * 64; i += 32) {
    int r = i >> 6, c = i & 63;
    int gr = row0 + r; if (gr >= n) gr = n - 1;   // clamp; garbage rows masked later
    t[r * TSTR + c] = X[(size_t)gr * 64 + c];
  }
}

// ---------------------------- small kernels ---------------------------------

// f32 row-major 64x64 -> bf16 swizzled fragment order
__global__ void k_convert(const float* s, bf16* d, int n) {
  int i = blockIdx.x * 256 + threadIdx.x;
  if (i < n) d[swz_idx(i >> 6, i & 63)] = (bf16)s[i];
}

__global__ void k_count(const int* src, const int* dst, float* cntL, float* cntR, int n) {
  int i = blockIdx.x * 256 + threadIdx.x;
  if (i < n) { atomicAdd(&cntL[src[i]], 1.f); atomicAdd(&cntR[dst[i]], 1.f); }
}

// count-weighted per-column sum / sumsq of h (n x 64)
__global__ void k_wstats(const float* h, const float* cnt, float* stats, int n) {
  __shared__ float s1[64], s2[64];
  int tid = threadIdx.x;
  if (tid < 64) { s1[tid] = 0.f; s2[tid] = 0.f; }
  __syncthreads();
  long total  = (long)n * 64;
  long stride = (long)gridDim.x * blockDim.x;      // multiple of 64 -> col fixed per thread
  long i0     = (long)blockIdx.x * blockDim.x + tid;
  int  col    = (int)(i0 & 63);
  float p = 0.f, q = 0.f;
  for (long i = i0; i < total; i += stride) {
    float w = cnt[i >> 6], x = h[i];
    p += w * x; q += w * x * x;
  }
  atomicAdd(&s1[col], p); atomicAdd(&s2[col], q);
  __syncthreads();
  if (tid < 64) { atomicAdd(&stats[tid], s1[tid]); atomicAdd(&stats[64 + tid], s2[tid]); }
}

__global__ void k_bn_finalize(const float* stats, float inv_denom, float* bn) {
  int i = threadIdx.x;                      // 64 threads
  float mu  = stats[i] * inv_denom;
  float var = stats[64 + i] * inv_denom - mu * mu;
  bn[i]      = mu;
  bn[64 + i] = rsqrtf(fmaxf(var, 0.f) + 1e-5f);
}

// Ws = diag(isd) @ W2 (bf16 swizzled, optional f32 row-major copy);
// c = b2 - (mu*isd) @ W2
__global__ void k_fold(const float* bn, const float* W2, const float* b2,
                       bf16* Ws, float* Wsf32, float* c) {
  int col = threadIdx.x;                    // 64 threads
  float acc = 0.f;
  for (int k = 0; k < 64; ++k) {
    float w = bn[64 + k] * W2[k * 64 + col];
    Ws[swz_idx(k, col)] = (bf16)w;
    if (Wsf32) Wsf32[k * 64 + col] = w;
    acc += bn[k] * w;
  }
  c[col] = b2[col] - acc;
}

// Wbc = Wb(f32 64x64) @ Wsf(f32 64x64) -> bf16 swizzled
__global__ void k_fold_bc(const float* Wb, const float* Wsf, bf16* out) {
  int t = blockIdx.x * 256 + threadIdx.x;
  if (t < 4096) {
    int r = t >> 6, c = t & 63;
    float a = 0.f;
    for (int k = 0; k < 64; ++k) a += Wb[r * 64 + k] * Wsf[k * 64 + c];
    out[swz_idx(r, c)] = (bf16)a;
  }
}

// ---------------------------- GEMM kernels ----------------------------------

// Y = act( X @ W (+bias) ),  X: n x 64 f32, W: 64x64 bf16 swizzled
__global__ void k_gemm_store(const float* X, const bf16* W, const float* bias, int relu,
                             float* Y, int n) {
  __shared__ float tile[WAVES][TROWS * TSTR];
  int wave = threadIdx.x >> 5, lane = threadIdx.x & 31;
  int row0 = (blockIdx.x * WAVES + wave) * TROWS;
  stage_raw(tile[wave], X, row0, n, lane);
  __syncthreads();
  v8f acc[4] = {};
  gemm_tile(tile[wave], W, lane, acc);
  int hi = (lane >> 4) & 1, c0 = lane & 15;
#pragma unroll
  for (int nt = 0; nt < 4; ++nt) {
    int col = c0 + nt * 16;
    float b = bias ? bias[col] : 0.f;
#pragma unroll
    for (int v = 0; v < 8; ++v) {
      int row = row0 + v + hi * 8;
      if (row < n) {
        float x = acc[nt][v] + b;
        if (relu) x = fmaxf(x, 0.f);
        Y[(size_t)row * 64 + col] = x;
      }
    }
  }
}

// P = relu( ((h-mu)*isd) @ W2 + b2 ) @ Wp   (per-node m_i/m_j -> Pi/Pj)
__global__ void k_node_chain(const float* h, const float* bn, const bf16* W2, const float* b2,
                             const bf16* Wp, float* P, int n) {
  __shared__ float tile[WAVES][TROWS * TSTR];
  int wave = threadIdx.x >> 5, lane = threadIdx.x & 31;
  int row0 = (blockIdx.x * WAVES + wave) * TROWS;
  float* t = tile[wave];
  for (int i = lane; i < TROWS * 64; i += 32) {         // stage with BN affine
    int r = i >> 6, c = i & 63;
    int gr = row0 + r; if (gr >= n) gr = n - 1;
    t[r * TSTR + c] = (h[(size_t)gr * 64 + c] - bn[c]) * bn[64 + c];
  }
  __syncthreads();
  v8f a1[4] = {};
  gemm_tile(t, W2, lane, a1);
  __syncthreads();
  d_to_lds(t, a1, lane, b2, 1);
  __syncthreads();
  v8f a2[4] = {};
  gemm_tile(t, Wp, lane, a2);
  int hi = (lane >> 4) & 1, c0 = lane & 15;
#pragma unroll
  for (int nt = 0; nt < 4; ++nt)
#pragma unroll
    for (int v = 0; v < 8; ++v) {
      int row = row0 + v + hi * 8;
      if (row < n) P[(size_t)row * 64 + c0 + nt * 16] = a2[nt][v];
    }
}

// pass 1: column stats of hE = edge @ We1 (raw, no bias -- BN cancels it)
__global__ void k_edge_pass1(const float* Xe, const bf16* W1, float* stats, int n) {
  __shared__ float tile[WAVES][TROWS * TSTR];
  __shared__ float s1[64], s2[64];
  int wave = threadIdx.x >> 5, lane = threadIdx.x & 31;
  if (threadIdx.x < 64) { s1[threadIdx.x] = 0.f; s2[threadIdx.x] = 0.f; }
  int row0 = (blockIdx.x * WAVES + wave) * TROWS;
  stage_raw(tile[wave], Xe, row0, n, lane);
  __syncthreads();
  v8f acc[4] = {};
  gemm_tile(tile[wave], W1, lane, acc);
  int hi = (lane >> 4) & 1, c0 = lane & 15;
#pragma unroll
  for (int nt = 0; nt < 4; ++nt) {
    int col = c0 + nt * 16;
    float p = 0.f, q = 0.f;
#pragma unroll
    for (int v = 0; v < 8; ++v) {
      int row = row0 + v + hi * 8;
      if (row < n) { float x = acc[nt][v]; p += x; q += x * x; }
    }
    atomicAdd(&s1[col], p); atomicAdd(&s2[col], q);
  }
  __syncthreads();
  if (threadIdx.x < 64) {
    atomicAdd(&stats[threadIdx.x], s1[threadIdx.x]);
    atomicAdd(&stats[64 + threadIdx.x], s2[threadIdx.x]);
  }
}

// pass 2: Pe = relu(hE @ We2s + ce) @ Wb; stats of h1 = Pi[dst]+Pe+Pj[src]
__global__ void k_edge_pass2(const float* Xe, const bf16* W1, const bf16* W2s, const float* ce,
                             const bf16* Wb, const float* Pi, const float* Pj,
                             const int* src, const int* dst, float* stats, int n) {
  __shared__ float tile[WAVES][TROWS * TSTR];
  __shared__ float s1[64], s2[64];
  int wave = threadIdx.x >> 5, lane = threadIdx.x & 31;
  if (threadIdx.x < 64) { s1[threadIdx.x] = 0.f; s2[threadIdx.x] = 0.f; }
  int row0 = (blockIdx.x * WAVES + wave) * TROWS;
  float* t = tile[wave];
  stage_raw(t, Xe, row0, n, lane);
  __syncthreads();
  v8f a1[4] = {};
  gemm_tile(t, W1, lane, a1);
  __syncthreads();
  d_to_lds(t, a1, lane, nullptr, 0);
  __syncthreads();
  v8f a2[4] = {};
  gemm_tile(t, W2s, lane, a2);
  __syncthreads();
  d_to_lds(t, a2, lane, ce, 1);                    // m_e tile
  __syncthreads();
  v8f a3[4] = {};
  gemm_tile(t, Wb, lane, a3);                      // Pe tile
  int hi = (lane >> 4) & 1, c0 = lane & 15;
#pragma unroll
  for (int nt = 0; nt < 4; ++nt) {
    int col = c0 + nt * 16;
    float p = 0.f, q = 0.f;
#pragma unroll
    for (int v = 0; v < 8; ++v) {
      int e = row0 + v + hi * 8;
      if (e < n) {
        int d = dst[e], s = src[e];
        float x = a3[nt][v] + Pi[(size_t)d * 64 + col] + Pj[(size_t)s * 64 + col];
        p += x; q += x * x;
      }
    }
    atomicAdd(&s1[col], p); atomicAdd(&s2[col], q);
  }
  __syncthreads();
  if (threadIdx.x < 64) {
    atomicAdd(&stats[threadIdx.x], s1[threadIdx.x]);
    atomicAdd(&stats[64 + threadIdx.x], s2[threadIdx.x]);
  }
}

// final: msg = relu(Qe + Qi[dst] + Qj[src] + c1); atomic segment-sum into agg[dst]
__global__ void k_edge_final(const float* Xe, const bf16* W1, const bf16* W2s, const float* ce,
                             const bf16* Wbc, const float* Qi, const float* Qj, const float* c1,
                             const int* src, const int* dst, float* agg, int n) {
  __shared__ float tile[WAVES][TROWS * TSTR];
  int wave = threadIdx.x >> 5, lane = threadIdx.x & 31;
  int row0 = (blockIdx.x * WAVES + wave) * TROWS;
  float* t = tile[wave];
  stage_raw(t, Xe, row0, n, lane);
  __syncthreads();
  v8f a1[4] = {};
  gemm_tile(t, W1, lane, a1);
  __syncthreads();
  d_to_lds(t, a1, lane, nullptr, 0);
  __syncthreads();
  v8f a2[4] = {};
  gemm_tile(t, W2s, lane, a2);
  __syncthreads();
  d_to_lds(t, a2, lane, ce, 1);
  __syncthreads();
  v8f a3[4] = {};
  gemm_tile(t, Wbc, lane, a3);                     // Qe tile
  int hi = (lane >> 4) & 1, c0 = lane & 15;
#pragma unroll
  for (int nt = 0; nt < 4; ++nt) {
    int col = c0 + nt * 16;
#pragma unroll
    for (int v = 0; v < 8; ++v) {
      int e = row0 + v + hi * 8;
      if (e < n) {
        int d = dst[e], s = src[e];
        float x = a3[nt][v] + Qi[(size_t)d * 64 + col] + Qj[(size_t)s * 64 + col] + c1[col];
        x = fmaxf(x, 0.f);
        atomicAdd(&agg[(size_t)d * 64 + col], x);
      }
    }
  }
}

// h_o = (agg/max(cnt,1)) @ U1 + right @ U2 (raw); store + per-column stats
__global__ void k_out1(const float* agg, const float* cnt, const float* right,
                       const bf16* U1, const bf16* U2, float* ho, float* stats, int n) {
  __shared__ float tile[WAVES][TROWS * TSTR];
  __shared__ float s1[64], s2[64];
  int wave = threadIdx.x >> 5, lane = threadIdx.x & 31;
  if (threadIdx.x < 64) { s1[threadIdx.x] = 0.f; s2[threadIdx.x] = 0.f; }
  int row0 = (blockIdx.x * WAVES + wave) * TROWS;
  float* t = tile[wave];
  for (int i = lane; i < TROWS * 64; i += 32) {        // stage normalized agg
    int r = i >> 6, c = i & 63;
    int gr = row0 + r; if (gr >= n) gr = n - 1;
    t[r * TSTR + c] = agg[(size_t)gr * 64 + c] / fmaxf(cnt[gr], 1.f);
  }
  __syncthreads();
  v8f acc[4] = {};
  gemm_tile(t, U1, lane, acc);
  __syncthreads();
  stage_raw(t, right, row0, n, lane);
  __syncthreads();
  gemm_tile(t, U2, lane, acc);                         // accumulate
  int hi = (lane >> 4) & 1, c0 = lane & 15;
#pragma unroll
  for (int nt = 0; nt < 4; ++nt) {
    int col = c0 + nt * 16;
    float p = 0.f, q = 0.f;
#pragma unroll
    for (int v = 0; v < 8; ++v) {
      int row = row0 + v + hi * 8;
      if (row < n) {
        float x = acc[nt][v];
        ho[(size_t)row * 64 + col] = x;
        p += x; q += x * x;
      }
    }
    atomicAdd(&s1[col], p); atomicAdd(&s2[col], q);
  }
  __syncthreads();
  if (threadIdx.x < 64) {
    atomicAdd(&stats[threadIdx.x], s1[threadIdx.x]);
    atomicAdd(&stats[64 + threadIdx.x], s2[threadIdx.x]);
  }
}

// ------------------------------- launcher -----------------------------------

extern "C" void kernel_launch(void* const* d_in, const int* in_sizes, int n_in,
                              void* d_out, int out_size, void* d_ws, size_t ws_size,
                              hipStream_t stream) {
  const float* left   = (const float*)d_in[0];
  const float* right  = (const float*)d_in[1];
  const float* edgef  = (const float*)d_in[2];
  const int*   ei     = (const int*)d_in[3];
  const float* fmr_w1 = (const float*)d_in[4];
  const float* fmr_w2 = (const float*)d_in[6];
  const float* fmr_b2 = (const float*)d_in[7];
  const float* fme_w1 = (const float*)d_in[8];
  const float* fme_w2 = (const float*)d_in[10];
  const float* fme_b2 = (const float*)d_in[11];
  const float* fmf_w1 = (const float*)d_in[12];
  const float* fmf_w2 = (const float*)d_in[14];
  const float* fmf_b2 = (const float*)d_in[15];
  const float* out_w1 = (const float*)d_in[16];
  const float* out_w2 = (const float*)d_in[18];
  const float* out_b2 = (const float*)d_in[19];

  int NL = in_sizes[0] / 64;
  int NR = in_sizes[1] / 64;
  int E  = in_sizes[2] / 64;
  const int* src = ei;
  const int* dst = ei + E;

  // ---- workspace carve (zero-init region first, one memset) ----
  float* f    = (float*)d_ws;
  float* cntR = f; f += NR;
  float* cntL = f; f += NL;
  float* agg  = f; f += (size_t)NR * 64;
  float* stats = f; f += 5 * 128;                  // I,J,E,1,O
  size_t zero_floats = (size_t)(f - (float*)d_ws);
  float* hR = f; f += (size_t)NR * 64;
  float* hL = f; f += (size_t)NL * 64;
  float* Pi = f; f += (size_t)NR * 64;
  float* Pj = f; f += (size_t)NL * 64;
  float* Qi = f; f += (size_t)NR * 64;
  float* Qj = f; f += (size_t)NL * 64;
  float* ho = f; f += (size_t)NR * 64;
  float* bn = f; f += 5 * 128;                     // (mu,isd) x 5
  float* ce = f; f += 64;
  float* c1 = f; f += 64;
  float* co = f; f += 64;
  float* Wsf = f; f += 4096;                       // f32 copy of fmf W2s
  // bf16 weight region, 64B aligned so 32B v16bf fragment loads are aligned
  uintptr_t wb = ((uintptr_t)f + 63) & ~(uintptr_t)63;
  bf16* w = (bf16*)wb;
  bf16 *Wr1 = w; w += 4096;  bf16 *Wr2 = w; w += 4096;
  bf16 *We1 = w; w += 4096;  bf16 *We2s = w; w += 4096;
  bf16 *Wa  = w; w += 4096;  bf16 *Wb   = w; w += 4096;
  bf16 *Wc  = w; w += 4096;  bf16 *Wbc  = w; w += 4096;
  bf16 *W2s = w; w += 4096;  bf16 *U1   = w; w += 4096;
  bf16 *U2  = w; w += 4096;  bf16 *Wo2s = w; w += 4096;

  hipMemsetAsync(d_ws, 0, zero_floats * sizeof(float), stream);

  // weight conversions (f32 row-major -> bf16 swizzled fragment order)
  k_convert<<<16, 256, 0, stream>>>(fmr_w1, Wr1, 4096);
  k_convert<<<16, 256, 0, stream>>>(fmr_w2, Wr2, 4096);
  k_convert<<<16, 256, 0, stream>>>(fme_w1, We1, 4096);
  k_convert<<<16, 256, 0, stream>>>(fmf_w1,        Wa, 4096);   // rows   0..63  (m_i)
  k_convert<<<16, 256, 0, stream>>>(fmf_w1 + 4096, Wb, 4096);   // rows  64..127 (m_e)
  k_convert<<<16, 256, 0, stream>>>(fmf_w1 + 8192, Wc, 4096);   // rows 128..191 (m_j)
  k_convert<<<16, 256, 0, stream>>>(out_w1,        U1, 4096);   // agg part
  k_convert<<<16, 256, 0, stream>>>(out_w1 + 4096, U2, 4096);   // right part

  k_count<<<(E + 255) / 256, 256, 0, stream>>>(src, dst, cntL, cntR, E);

  int gNR = (NR + WAVES * TROWS - 1) / (WAVES * TROWS);
  int gNL = (NL + WAVES * TROWS - 1) / (WAVES * TROWS);
  int gE  = (E  + WAVES * TROWS - 1) / (WAVES * TROWS);

  // per-node first linears (bias dropped: BN cancels it)
  k_gemm_store<<<gNR, 256, 0, stream>>>(right, Wr1, nullptr, 0, hR, NR);
  k_gemm_store<<<gNL, 256, 0, stream>>>(left,  Wr1, nullptr, 0, hL, NL);

  // count-weighted BN stats for m_i / m_j
  k_wstats<<<256, 256, 0, stream>>>(hR, cntR, stats + 0,   NR);
  k_wstats<<<256, 256, 0, stream>>>(hL, cntL, stats + 128, NL);
  k_bn_finalize<<<1, 64, 0, stream>>>(stats + 0,   1.f / (float)E, bn + 0);
  k_bn_finalize<<<1, 64, 0, stream>>>(stats + 128, 1.f / (float)E, bn + 128);

  // Pi = m_i @ Wa, Pj = m_j @ Wc
  k_node_chain<<<gNR, 256, 0, stream>>>(hR, bn + 0,   Wr2, fmr_b2, Wa, Pi, NR);
  k_node_chain<<<gNL, 256, 0, stream>>>(hL, bn + 128, Wr2, fmr_b2, Wc, Pj, NL);

  // edge block: stats of hE, fold BN into fme_w2
  k_edge_pass1<<<gE, 256, 0, stream>>>(edgef, We1, stats + 256, E);
  k_bn_finalize<<<1, 64, 0, stream>>>(stats + 256, 1.f / (float)E, bn + 256);
  k_fold<<<1, 64, 0, stream>>>(bn + 256, fme_w2, fme_b2, We2s, nullptr, ce);

  // h1 stats (Pi[dst] + Pe + Pj[src]), fold BN into fmf_w2, combine Wb@W2s
  k_edge_pass2<<<gE, 256, 0, stream>>>(edgef, We1, We2s, ce, Wb, Pi, Pj, src, dst,
                                       stats + 384, E);
  k_bn_finalize<<<1, 64, 0, stream>>>(stats + 384, 1.f / (float)E, bn + 384);
  k_fold<<<1, 64, 0, stream>>>(bn + 384, fmf_w2, fmf_b2, W2s, Wsf, c1);
  k_fold_bc<<<16, 256, 0, stream>>>(fmf_w1 + 4096, Wsf, Wbc);

  // Qi = Pi @ W2s, Qj = Pj @ W2s
  k_gemm_store<<<gNR, 256, 0, stream>>>(Pi, W2s, nullptr, 0, Qi, NR);
  k_gemm_store<<<gNL, 256, 0, stream>>>(Pj, W2s, nullptr, 0, Qj, NL);

  // messages + mean-aggregation numerator
  k_edge_final<<<gE, 256, 0, stream>>>(edgef, We1, We2s, ce, Wbc, Qi, Qj, c1,
                                       src, dst, agg, E);

  // output block
  k_out1<<<gNR, 256, 0, stream>>>(agg, cntR, right, U1, U2, ho, stats + 512, NR);
  k_bn_finalize<<<1, 64, 0, stream>>>(stats + 512, 1.f / (float)NR, bn + 512);
  k_fold<<<1, 64, 0, stream>>>(bn + 512, out_w2, out_b2, Wo2s, nullptr, co);
  k_gemm_store<<<gNR, 256, 0, stream>>>(ho, Wo2s, co, 1, (float*)d_out, NR);

  (void)n_in; (void)out_size; (void)ws_size;
}
